// RGCNHighMem_3908420239947
// MI455X (gfx1250) — matrix operand order repaired and checked by reading, not compiled, not run on hardware
//
#include <hip/hip_runtime.h>

typedef __attribute__((ext_vector_type(16))) _Float16 v16h;
typedef __attribute__((ext_vector_type(8)))  float    v8f;

#define D        32
#define NRELS    64

// ws layout (bytes):
//   0      : counts[64]  (int)
//   1024   : offsets[65] (int)
//   2048   : cursors[64] (int)
//   4096   : packedW[NRELS*2*32*16] _Float16  (128 KB), B-operand lane layout
//   135168 : edge_sd[E]  (int2: {src, dst} of edge, bucketed by relation)
#define WS_COUNTS   0
#define WS_OFFSETS  1024
#define WS_CURSORS  2048
#define WS_PACKED   4096
#define WS_EDGESD   (WS_PACKED + NRELS * 2 * 32 * 16 * 2)

// ---------------------------------------------------------------- zero
__global__ void rgcn_zero_kernel(float* __restrict__ out, int n,
                                 int* __restrict__ counts) {
    int i = blockIdx.x * blockDim.x + threadIdx.x;
    for (; i < n; i += gridDim.x * blockDim.x) out[i] = 0.0f;
    int t = blockIdx.x * blockDim.x + threadIdx.x;
    if (t < NRELS) counts[t] = 0;
}

// ------------------------------------------------- pack W -> f16 B layout
// B (K x N = 32 x 16) operand layout used here (contiguous k, same order as A), wave32:
//   lane l: n = l & 15, kbase = (l >= 16) ? 16 : 0, element j -> K = kbase + j
__global__ void rgcn_packw_kernel(const float* __restrict__ weight,
                                  _Float16* __restrict__ packed) {
    int g = blockIdx.x * blockDim.x + threadIdx.x;
    if (g >= NRELS * 2 * 32 * 16) return;
    int j    = g & 15;
    int lane = (g >> 4) & 31;
    int mat  = (g >> 9) & 1;
    int r    = g >> 10;
    int k    = ((lane & 16) ? 16 : 0) + j;
    int col  = mat * 16 + (lane & 15);
    packed[g] = (_Float16)weight[((size_t)r * D + k) * D + col];
}

// ---------------------------------------------------------------- histogram
__global__ void rgcn_hist_kernel(const int* __restrict__ etypes, int E,
                                 int* __restrict__ counts) {
    int i = blockIdx.x * blockDim.x + threadIdx.x;
    for (; i < E; i += gridDim.x * blockDim.x)
        atomicAdd(&counts[etypes[i]], 1);
}

// ---------------------------------------------------------------- scan (64 elems)
__global__ void rgcn_scan_kernel(const int* __restrict__ counts,
                                 int* __restrict__ offsets,
                                 int* __restrict__ cursors) {
    if (threadIdx.x == 0 && blockIdx.x == 0) {
        int acc = 0;
        for (int r = 0; r < NRELS; ++r) {
            offsets[r] = acc;
            cursors[r] = acc;
            acc += counts[r];
        }
        offsets[NRELS] = acc;
    }
}

// ------------------------------------------- bucket edges: store {src,dst}
__global__ void rgcn_scatter_kernel(const int* __restrict__ etypes,
                                    const int* __restrict__ src,
                                    const int* __restrict__ dst, int E,
                                    int* __restrict__ cursors,
                                    int2* __restrict__ edge_sd) {
    int i = blockIdx.x * blockDim.x + threadIdx.x;
    for (; i < E; i += gridDim.x * blockDim.x) {
        int pos = atomicAdd(&cursors[etypes[i]], 1);
        edge_sd[pos] = make_int2(src[i], dst[i]);
    }
}

// ---------------------------------------------------------------- WMMA compute
// grid = (NRELS, SLICES), block = 128 (4 waves). Wave processes 16-edge tiles
// of relation blockIdx.x, strided across slices*waves.
__global__ void __launch_bounds__(128)
rgcn_wmma_kernel(const float* __restrict__ feat,
                 const int2* __restrict__ edge_sd,
                 const int* __restrict__ offsets,
                 const _Float16* __restrict__ packedW,
                 float* __restrict__ out) {
    const int lane = threadIdx.x & 31;
    const int wave = threadIdx.x >> 5;
    const int r    = blockIdx.x;

    const int seg_begin = offsets[r];
    const int seg_end   = offsets[r + 1];
    const int count     = seg_end - seg_begin;
    if (count <= 0) return;
    const int ntiles = (count + 15) >> 4;

    // Per-relation B operands, resident in VGPRs for the whole loop.
    const v16h b0 = *(const v16h*)(packedW + ((size_t)(r * 2 + 0) * 32 + lane) * 16);
    const v16h b1 = *(const v16h*)(packedW + ((size_t)(r * 2 + 1) * 32 + lane) * 16);

    const int m   = lane & 15;          // A row owned by this lane / D column
    const int hi  = (lane >> 4) & 1;    // lane >= 16
    const int kbA = hi ? 16 : 0;        // A layout: lo lanes K 0-15, hi lanes K 16-31

    const int wavesPerRel = gridDim.y * (blockDim.x >> 5);
    for (int t = blockIdx.y * (blockDim.x >> 5) + wave; t < ntiles; t += wavesPerRel) {
        const int tile_base = seg_begin + (t << 4);
        const int rem  = seg_end - tile_base;
        // mcnt is wave-uniform; force it into an SGPR so the full/tail-tile
        // split becomes a scalar branch (no EXEC-mask dance on the hot path).
        const int mcnt = __builtin_amdgcn_readfirstlane(rem < 16 ? rem : 16);
        const bool valid = (m < mcnt);

        // One b64 load gives this lane's row: {src node, dst node}.
        const int2 sd = edge_sd[tile_base + (valid ? m : 0)];
        const float* row = feat + (size_t)sd.x * D;

        float4 f0 = *(const float4*)(row + kbA);
        float4 f1 = *(const float4*)(row + kbA + 4);
        float4 f2 = *(const float4*)(row + kbA + 8);
        float4 f3 = *(const float4*)(row + kbA + 12);
        const float zmask = valid ? 1.0f : 0.0f;

        v16h a;
        a[0]  = (_Float16)(f0.x * zmask); a[1]  = (_Float16)(f0.y * zmask);
        a[2]  = (_Float16)(f0.z * zmask); a[3]  = (_Float16)(f0.w * zmask);
        a[4]  = (_Float16)(f1.x * zmask); a[5]  = (_Float16)(f1.y * zmask);
        a[6]  = (_Float16)(f1.z * zmask); a[7]  = (_Float16)(f1.w * zmask);
        a[8]  = (_Float16)(f2.x * zmask); a[9]  = (_Float16)(f2.y * zmask);
        a[10] = (_Float16)(f2.z * zmask); a[11] = (_Float16)(f2.w * zmask);
        a[12] = (_Float16)(f3.x * zmask); a[13] = (_Float16)(f3.y * zmask);
        a[14] = (_Float16)(f3.z * zmask); a[15] = (_Float16)(f3.w * zmask);

        v8f c0 = {};
        v8f c1 = {};
        c0 = __builtin_amdgcn_wmma_f32_16x16x32_f16(false, a, false, b0,
                                                    (short)0, c0, false, false);
        c1 = __builtin_amdgcn_wmma_f32_16x16x32_f16(false, a, false, b1,
                                                    (short)0, c1, false, false);

        // Broadcast dst of row (rr + 8*hi) from the lane that owns it.
        // Done unconditionally with full EXEC (shuffle sources lanes 0-15).
        int d8[8];
#pragma unroll
        for (int rr = 0; rr < 8; ++rr)
            d8[rr] = __shfl(sd.y, rr + (hi ? 8 : 0), 32);

        // D layout: lane -> col (lane&15), reg rr -> row rr + 8*hi
        float* const outc = out + m;   // column base for this lane
        if (mcnt == 16) {
            // Full tile (scalar branch): branch-free run of 16 fp atomics.
#pragma unroll
            for (int rr = 0; rr < 8; ++rr) {
                atomicAdd(&outc[(size_t)d8[rr] * D],      c0[rr]);
                atomicAdd(&outc[(size_t)d8[rr] * D + 16], c1[rr]);
            }
        } else {
#pragma unroll
            for (int rr = 0; rr < 8; ++rr) {
                const int mm = rr + (hi ? 8 : 0);
                if (mm < mcnt) {
                    atomicAdd(&outc[(size_t)d8[rr] * D],      c0[rr]);
                    atomicAdd(&outc[(size_t)d8[rr] * D + 16], c1[rr]);
                }
            }
        }
    }
}

// ---------------------------------------------------------------- launch
extern "C" void kernel_launch(void* const* d_in, const int* in_sizes, int n_in,
                              void* d_out, int out_size, void* d_ws, size_t ws_size,
                              hipStream_t stream) {
    const float* feat   = (const float*)d_in[0];
    const float* weight = (const float*)d_in[1];
    const int*   src    = (const int*)d_in[2];
    const int*   dst    = (const int*)d_in[3];
    const int*   etypes = (const int*)d_in[4];
    float*       out    = (float*)d_out;
    const int E = in_sizes[2];

    char* ws = (char*)d_ws;
    int*      counts  = (int*)(ws + WS_COUNTS);
    int*      offsets = (int*)(ws + WS_OFFSETS);
    int*      cursors = (int*)(ws + WS_CURSORS);
    _Float16* packedW = (_Float16*)(ws + WS_PACKED);
    int2*     edge_sd = (int2*)(ws + WS_EDGESD);

    // 1) zero output + counters
    {
        int n = out_size;
        int blocks = (n + 255) / 256;
        if (blocks > 2048) blocks = 2048;
        rgcn_zero_kernel<<<blocks, 256, 0, stream>>>(out, n, counts);
    }
    // 2) pack weights into f16 B-operand layout
    {
        int n = NRELS * 2 * 32 * 16;
        rgcn_packw_kernel<<<(n + 255) / 256, 256, 0, stream>>>(weight, packedW);
    }
    // 3) histogram per relation
    {
        int blocks = (E + 255) / 256;
        if (blocks > 2048) blocks = 2048;
        rgcn_hist_kernel<<<blocks, 256, 0, stream>>>(etypes, E, counts);
    }
    // 4) exclusive scan over 64 relations
    rgcn_scan_kernel<<<1, 64, 0, stream>>>(counts, offsets, cursors);
    // 5) bucket edges by relation, storing {src,dst} records
    {
        int blocks = (E + 255) / 256;
        if (blocks > 2048) blocks = 2048;
        rgcn_scatter_kernel<<<blocks, 256, 0, stream>>>(etypes, src, dst, E,
                                                        cursors, edge_sd);
    }
    // 6) WMMA edge-tile compute + scatter-add
    {
        dim3 grid(NRELS, 8);
        rgcn_wmma_kernel<<<grid, 128, 0, stream>>>(feat, edge_sd, offsets,
                                                   packedW, out);
    }
}
